// InfoNCELayer_11708080849216
// MI455X (gfx1250) — matrix operand order repaired and checked by reading, not compiled, not run on hardware
//
#include <hip/hip_runtime.h>
#include <hip/hip_bf16.h>
#include <math.h>

// ---------------------------------------------------------------------------
// Problem constants (from reference): T=16, B=32, C=O=256, H=W=16 (HW=256), K=4
// ---------------------------------------------------------------------------
#define T_DIM  16
#define B_DIM  32
#define C_DIM  256
#define HW_DIM 256
#define KLAG   4
#define TP     (T_DIM - KLAG)            // 12 usable timesteps
#define CHW    (C_DIM * HW_DIM)          // 65536 elements per (t,b) slab
#define SITES  (B_DIM * CHW)             // 2097152 sites per timestep
#define N_SPK  ((size_t)T_DIM * SITES)   // 33554432 floats (output 0)

typedef __bf16 bf16_t;
typedef __attribute__((ext_vector_type(16))) __bf16 v16bf;
typedef __attribute__((ext_vector_type(8)))  __bf16 v8bf;
typedef __attribute__((ext_vector_type(8)))  float  v8f;

// ---------------------------------------------------------------------------
// Kernel 1: leaky-integrate-fire scan (sequential over T per site) + pass-
// through copy of spk_rec into d_out.  context[t] (t<12) stored as bf16
// (values are exactly 0/1 -> lossless) for the WMMA consumer.
// ---------------------------------------------------------------------------
__global__ __launch_bounds__(256)
void scan_copy_kernel(const float* __restrict__ spk,
                      float* __restrict__ out,
                      bf16_t* __restrict__ ctx) {
    const int site = blockIdx.x * blockDim.x + threadIdx.x;
    if (site >= SITES) return;
    float mem = 0.0f;
#pragma unroll
    for (int t = 0; t < T_DIM; ++t) {
        const float inp = spk[(size_t)t * SITES + site];
        out[(size_t)t * SITES + site] = inp;           // output 0 pass-through
        const float reset = (mem - 1.0f > 0.0f) ? 1.0f : 0.0f;
        mem = 0.5f * mem + inp - reset;                // BETA=0.5, THRESH=1.0
        const float s = (mem - 1.0f > 0.0f) ? 1.0f : 0.0f;
        if (t < TP) ctx[(size_t)t * SITES + site] = (bf16_t)s;
    }
}

// ---------------------------------------------------------------------------
// Kernel 2: split f32 weight into bf16 hi + bf16 lo (W = Whi + Wlo) so two
// bf16 WMMAs reproduce ~f32 GEMM accuracy (A/B spikes are exact 0/1 in bf16).
// ---------------------------------------------------------------------------
__global__ __launch_bounds__(256)
void wsplit_kernel(const float* __restrict__ w,
                   bf16_t* __restrict__ whi,
                   bf16_t* __restrict__ wlo) {
    const int i = blockIdx.x * blockDim.x + threadIdx.x;
    if (i >= C_DIM * C_DIM) return;
    const float v = w[i];
    const bf16_t h = (bf16_t)v;
    whi[i] = h;
    wlo[i] = (bf16_t)(v - (float)h);
}

// ---------------------------------------------------------------------------
// Kernel 3: inverse permutation of rand_idx (x_neg[t,b] = x_pos[t,rand_idx[b]]
// => block owning batch b' writes neg(t, inv[b'])).
// ---------------------------------------------------------------------------
__global__ void invperm_kernel(const int* __restrict__ ridx,
                               int* __restrict__ inv) {
    const int b = threadIdx.x;
    if (b < B_DIM) inv[ridx[b]] = b;
}

// ---------------------------------------------------------------------------
// Kernel 4: fused GEMM + dual context reduction.
// One block per (t', b') pair, 12*32 = 384 blocks, 256 threads = 8 wave32s.
//   x[o,hw] = sum_c (Whi+Wlo)[o,c] * spk[t'+4, b', c, hw]   (v_wmma bf16)
//   pos(t',b')        = sum_{o,hw} ctx[t',b',o,hw]      * x[o,hw] / 65536
//   neg(t',inv[b'])   = sum_{o,hw} ctx[t',inv[b'],o,hw] * x[o,hw] / 65536
// spk tile staged into LDS transposed ([hw][c], padded stride) with on-the-fly
// f32->bf16 conversion so B fragments are contiguous 16B LDS loads per lane.
// ---------------------------------------------------------------------------
#define LDS_STRIDE 264   // bf16 elems per hw-row: 528B = 33*16 -> 16B aligned

__global__ __launch_bounds__(256)
void gemm_reduce_kernel(const float* __restrict__ spk,
                        const bf16_t* __restrict__ ctx,
                        const bf16_t* __restrict__ whi,
                        const bf16_t* __restrict__ wlo,
                        const int* __restrict__ inv,
                        float* __restrict__ posm,
                        float* __restrict__ negm) {
    extern __shared__ char smem_raw[];
    bf16_t* lds_b = (bf16_t*)smem_raw;                         // [256][264]
    float*  red   = (float*)(smem_raw + HW_DIM * LDS_STRIDE * sizeof(bf16_t));

    const int tp   = blockIdx.x >> 5;    // 0..11
    const int b    = blockIdx.x & 31;    // 0..31
    const int tid  = threadIdx.x;
    const int lane = tid & 31;
    const int wave = tid >> 5;

    // ---- stage transposed bf16 spike tile into LDS --------------------------
    {
        const float* spk_src = spk + ((size_t)(tp + KLAG) * B_DIM + b) * CHW;
        const int hw = tid;  // blockDim.x == 256 == HW_DIM
#pragma unroll 4
        for (int c = 0; c < C_DIM; ++c) {
            const float v = spk_src[(size_t)c * HW_DIM + hw];  // coalesced
            lds_b[hw * LDS_STRIDE + c] = (bf16_t)v;
        }
    }

    const int bneg = inv[b];
    const bf16_t* ctxPos = ctx + ((size_t)tp * B_DIM + b)    * CHW;
    const bf16_t* ctxNeg = ctx + ((size_t)tp * B_DIM + bneg) * CHW;

    __syncthreads();

    float posAcc = 0.0f, negAcc = 0.0f;

    // WMMA fragment lane geometry (wave32, 16x16x32 bf16):
    //  A (16x32): lane holds row o = l16;  K = cb + hi8 + [0..7]  (elems 0..7)
    //                                      K = cb + 16 + hi8 + [0..7] (8..15)
    //  B (32x16): lane holds col hw = l16; K same split as A (hi8, 16+hi8)
    //  C/D: VGPR j, lanes0-15: M=j N=lane; lanes16-31: M=j+8 N=lane-16
    const int l16  = lane & 15;
    const int hiG  = lane >> 4;          // 0 | 1
    const int hi8  = hiG << 3;           // 0 | 8
    const int hi16 = hiG << 4;           // 0 | 16

    for (int ot = wave; ot < 16; ot += 8) {            // 2 o-tiles per wave
        const int o_row = ot * 16 + l16;
        const bf16_t* whi_row = whi + o_row * C_DIM;
        const bf16_t* wlo_row = wlo + o_row * C_DIM;
        for (int ht = 0; ht < 16; ++ht) {              // 16 hw-tiles
            v8f acc = {};
            const bf16_t* bbase = lds_b + (ht * 16 + l16) * LDS_STRIDE + hi8;
#pragma unroll
            for (int cc = 0; cc < 8; ++cc) {           // K = 256 in 8 steps
                const int cb = cc * 32;
                // B fragment: two 16B chunks (c = cb+hi8, cb+16+hi8) of a transposed LDS row
                v8bf b0 = *(const v8bf*)(bbase + cb);
                v8bf b1 = *(const v8bf*)(bbase + cb + 16);
                v16bf bm = __builtin_shufflevector(b0, b1,
                    0,1,2,3,4,5,6,7,8,9,10,11,12,13,14,15);
                // A hi fragment: two 16B chunks of a weight row (L2 resident)
                v8bf a0 = *(const v8bf*)(whi_row + cb + hi8);
                v8bf a1 = *(const v8bf*)(whi_row + cb + 16 + hi8);
                v16bf ah = __builtin_shufflevector(a0, a1,
                    0,1,2,3,4,5,6,7,8,9,10,11,12,13,14,15);
                acc = __builtin_amdgcn_wmma_f32_16x16x32_bf16(
                        false, ah, false, bm, (short)0, acc, false, false);
                // A lo fragment (error-correction term)
                v8bf l0 = *(const v8bf*)(wlo_row + cb + hi8);
                v8bf l1 = *(const v8bf*)(wlo_row + cb + 16 + hi8);
                v16bf al = __builtin_shufflevector(l0, l1,
                    0,1,2,3,4,5,6,7,8,9,10,11,12,13,14,15);
                acc = __builtin_amdgcn_wmma_f32_16x16x32_bf16(
                        false, al, false, bm, (short)0, acc, false, false);
            }
            // fused elementwise reduction against both context matrices
            const int hw = ht * 16 + l16;
#pragma unroll
            for (int j = 0; j < 8; ++j) {
                const int o   = ot * 16 + j + hi8;
                const int idx = o * HW_DIM + hw;
                posAcc = fmaf((float)ctxPos[idx], acc[j], posAcc);
                negAcc = fmaf((float)ctxNeg[idx], acc[j], negAcc);
            }
        }
    }

    // ---- block reductions (reuse LDS tail) ---------------------------------
    red[tid] = posAcc;
    __syncthreads();
    for (int s = 128; s > 0; s >>= 1) {
        if (tid < s) red[tid] += red[tid + s];
        __syncthreads();
    }
    if (tid == 0) posm[tp * B_DIM + b] = red[0] * (1.0f / 65536.0f);
    __syncthreads();
    red[tid] = negAcc;
    __syncthreads();
    for (int s = 128; s > 0; s >>= 1) {
        if (tid < s) red[tid] += red[tid + s];
        __syncthreads();
    }
    if (tid == 0) negm[tp * B_DIM + bneg] = red[0] * (1.0f / 65536.0f);
}

// ---------------------------------------------------------------------------
// Kernel 5: scalar epilogue (384 scores) -> loss, log_score_pos, log_score_neg
// Output layout: [spk_rec (33554432)] [loss (1)] [lsp (384)] [lsn (12)]
// ---------------------------------------------------------------------------
__global__ __launch_bounds__(512)
void finalize_kernel(const float* __restrict__ posm,
                     const float* __restrict__ negm,
                     float* __restrict__ out) {
    __shared__ float slsp[TP * B_DIM];
    __shared__ float ssneg[TP * B_DIM];
    __shared__ float slsn[TP];
    const int tid = threadIdx.x;
    if (tid < TP * B_DIM) {
        const float lsp = logf(expf(posm[tid]) + 1.0e-4f);
        slsp[tid] = lsp;
        out[N_SPK + 1 + tid] = lsp;                    // log_score_pos
        ssneg[tid] = expf(negm[tid]) + 1.0e-4f;        // score_neg
    }
    __syncthreads();
    if (tid < TP) {
        float s = 0.0f;
        for (int bb = 0; bb < B_DIM; ++bb) s += ssneg[tid * B_DIM + bb];
        const float lsn = logf(s);
        slsn[tid] = lsn;
        out[N_SPK + 1 + TP * B_DIM + tid] = lsn;       // log_score_neg
    }
    __syncthreads();
    if (tid == 0) {
        float L = 0.0f;
        for (int i = 0; i < TP * B_DIM; ++i) L += (slsn[i / B_DIM] - slsp[i]);
        out[N_SPK] = L * (1.0f / (float)(TP * B_DIM)); // loss
    }
}

// ---------------------------------------------------------------------------
// Workspace layout (bytes):
//   [0, 50331648)          ctx bf16    : 12 * SITES * 2
//   [50331648, 50462720)   Whi bf16    : 65536 * 2
//   [50462720, 50593792)   Wlo bf16    : 65536 * 2
//   [50593792, 50595328)   pos means   : 384 * 4
//   [50595328, 50596864)   neg means   : 384 * 4
//   [50596864, 50596992)   inv perm    : 32 * 4
// ---------------------------------------------------------------------------
extern "C" void kernel_launch(void* const* d_in, const int* in_sizes, int n_in,
                              void* d_out, int out_size, void* d_ws, size_t ws_size,
                              hipStream_t stream) {
    const float* spk  = (const float*)d_in[0];
    /* d_in[1] (mem_rec) is unused by the reference computation */
    const float* w    = (const float*)d_in[2];
    const int*   ridx = (const int*)d_in[3];
    float* out = (float*)d_out;
    char*  ws  = (char*)d_ws;

    bf16_t* ws_ctx = (bf16_t*)(ws);
    bf16_t* ws_whi = (bf16_t*)(ws + 50331648);
    bf16_t* ws_wlo = (bf16_t*)(ws + 50462720);
    float*  ws_pos = (float*)(ws + 50593792);
    float*  ws_neg = (float*)(ws + 50595328);
    int*    ws_inv = (int*)(ws + 50596864);

    scan_copy_kernel<<<SITES / 256, 256, 0, stream>>>(spk, out, ws_ctx);
    wsplit_kernel<<<(C_DIM * C_DIM) / 256, 256, 0, stream>>>(w, ws_whi, ws_wlo);
    invperm_kernel<<<1, 32, 0, stream>>>(ridx, ws_inv);

    const size_t smem = (size_t)HW_DIM * LDS_STRIDE * sizeof(bf16_t)
                      + 256 * sizeof(float);          // 136,192 B (<320KB WGP LDS)
    gemm_reduce_kernel<<<TP * B_DIM, 256, smem, stream>>>(
        spk, ws_ctx, ws_whi, ws_wlo, ws_inv, ws_pos, ws_neg);

    finalize_kernel<<<1, 512, 0, stream>>>(ws_pos, ws_neg, out);
}
